// ManifoldAttentionNoAttnStage2_75522704933231
// MI455X (gfx1250) — matrix operand order repaired and checked by reading, not compiled, not run on hardware
//
#include <hip/hip_runtime.h>
#include <cstdint>
#include <cstddef>

#define DEV __device__ __forceinline__

static constexpr int   Bn  = 8, Tn = 4096, Dn = 1024, Rn = 16, KIT = 4, SRn = 128;
static constexpr float EPSc = 1e-5f, LAMc = 0.02f, LNEPSc = 1e-5f;

typedef __bf16 bf16_t;
typedef bf16_t v16bf __attribute__((ext_vector_type(16)));
typedef bf16_t v8bf  __attribute__((ext_vector_type(8)));
typedef float  v8f   __attribute__((ext_vector_type(8)));
typedef float  v4f   __attribute__((ext_vector_type(4)));

DEV float gelu_erf(float x) { return 0.5f * x * (1.0f + erff(x * 0.70710678118654752f)); }
DEV float soft_shrink(float x) {
    float s = (x > 0.f) ? 1.f : ((x < 0.f) ? -1.f : 0.f);
    return s * gelu_erf(fabsf(x) - LAMc);
}
DEV v8f vzero8() { v8f z = {0.f,0.f,0.f,0.f,0.f,0.f,0.f,0.f}; return z; }

// A-operand (16x32 bf16): lane L -> row = L&15, hi = L>>4.
// VGPR0-7 hold K = hi*16 .. hi*16+15.
DEV v16bf wmma_lda_bf(const bf16_t* rowk, int hi) {
    v8bf a0 = *(const v8bf*)(rowk + hi * 16);
    v8bf a1 = *(const v8bf*)(rowk + hi * 16 + 8);
    v16bf r;
#pragma unroll
    for (int i = 0; i < 8; ++i) { r[i] = a0[i]; r[i + 8] = a1[i]; }
    return r;
}
// A-operand built from fp32 source (on-the-fly cvt)
DEV v16bf wmma_lda_f32(const float* rowk, int hi) {
    v4f f0 = *(const v4f*)(rowk + hi * 16);
    v4f f1 = *(const v4f*)(rowk + hi * 16 + 4);
    v4f f2 = *(const v4f*)(rowk + hi * 16 + 8);
    v4f f3 = *(const v4f*)(rowk + hi * 16 + 12);
    v16bf r;
#pragma unroll
    for (int i = 0; i < 4; ++i) {
        r[i]      = (bf16_t)f0[i];
        r[i + 4]  = (bf16_t)f1[i];
        r[i + 8]  = (bf16_t)f2[i];
        r[i + 12] = (bf16_t)f3[i];
    }
    return r;
}
DEV v8f wmma_bf16(v16bf a, v16bf b, v8f c) {
    return __builtin_amdgcn_wmma_f32_16x16x32_bf16(false, a, false, b, (short)0, c, false, false);
}

// ---- gfx1250 async Global->LDS DMA (tracked by ASYNCcnt) --------------------
// LDS destination operand is the raw LDS byte offset: generic LDS pointers are
// {aperture_hi32, lds_offset_lo32} (ISA 10.2), so truncation yields the offset.
DEV unsigned lds_off32(const void* p) { return (unsigned)(size_t)p; }
DEV void async_copy_b128(unsigned lds_byte_off, const void* gptr) {
    asm volatile("global_load_async_to_lds_b128 %0, %1, off"
                 :: "v"(lds_byte_off), "v"(gptr) : "memory");
}
DEV void wait_async0() { asm volatile("s_wait_asynccnt 0" ::: "memory"); }

DEV float block_sum(float v, float* red) {
#pragma unroll
    for (int o = 16; o > 0; o >>= 1) v += __shfl_down(v, o, 32);
    int lane = threadIdx.x & 31, w = threadIdx.x >> 5;
    if (lane == 0) red[w] = v;
    __syncthreads();
    if (threadIdx.x == 0) { float s = 0.f; for (int i = 0; i < 8; ++i) s += red[i]; red[0] = s; }
    __syncthreads();
    float s = red[0];
    __syncthreads();
    return s;
}

// ---------------------------------------------------------------- utilities
__global__ void k_cvt_bf16(const float* __restrict__ src, bf16_t* __restrict__ dst, int n) {
    int i = blockIdx.x * 256 + threadIdx.x;
    if (i < n) dst[i] = (bf16_t)src[i];
}

__global__ void k_init_V(float* __restrict__ V, bf16_t* __restrict__ Vt) {
    int i = blockIdx.x * 256 + threadIdx.x;
    if (i >= Bn * Dn * Rn) return;
    int r = i % Rn, d = (i / Rn) % Dn, b = i / (Rn * Dn);
    float v = (d == r) ? 1.f : 0.f;
    V[i] = v;
    Vt[((size_t)b * Rn + r) * Dn + d] = (bf16_t)v;
}

// ------------------------------------------- low-rank shift + xprime (WMMA)
// h = gelu(x @ Win^T) [16x128];  s = h @ Wout^T + b;  xprime = x - e1 + s
__global__ __launch_bounds__(256) void k_shift(
    const float* __restrict__ x, const bf16_t* __restrict__ Winb,
    const bf16_t* __restrict__ Woutb, const float* __restrict__ sob,
    float* __restrict__ xprime, bf16_t* __restrict__ xpb)
{
    __shared__ bf16_t hs[8][16 * SRn];  // 4KB per wave
    int lane = threadIdx.x & 31, w = threadIdx.x >> 5;
    int tile = blockIdx.x * 8 + w;
    int b = tile / (Tn / 16);
    int t0 = (tile % (Tn / 16)) * 16;
    int nrow = lane & 15, hi = lane >> 4;

    const float* xrow = x + ((size_t)b * Tn + t0 + nrow) * Dn;

    v8f acc[8];
#pragma unroll
    for (int j = 0; j < 8; ++j) acc[j] = vzero8();

    for (int kk = 0; kk < Dn; kk += 32) {
        v16bf A = wmma_lda_f32(xrow + kk, hi);
#pragma unroll
        for (int j = 0; j < 8; ++j) {
            const bf16_t* bp = Winb + (size_t)(j * 16 + nrow) * Dn + kk + hi * 16;
            acc[j] = wmma_bf16(A, *(const v16bf*)bp, acc[j]);
        }
    }
    // gelu + stash h in LDS (C-layout scatter: lane col = nrow, rows i+8*hi)
#pragma unroll
    for (int j = 0; j < 8; ++j)
#pragma unroll
        for (int i = 0; i < 8; ++i) {
            int m = i + 8 * hi;
            hs[w][m * SRn + j * 16 + nrow] = (bf16_t)gelu_erf(acc[j][i]);
        }
    __syncthreads();

    const bf16_t* hrow = &hs[w][nrow * SRn];
    for (int jn = 0; jn < Dn / 16; ++jn) {
        v8f c = vzero8();
        int d = jn * 16 + nrow;
#pragma unroll
        for (int kk = 0; kk < SRn; kk += 32) {
            v16bf A = wmma_lda_bf(hrow + kk, hi);
            const bf16_t* bp = Woutb + (size_t)d * SRn + kk + hi * 16;
            c = wmma_bf16(A, *(const v16bf*)bp, c);
        }
        float bias = sob[d];
#pragma unroll
        for (int i = 0; i < 8; ++i) {
            int t = t0 + i + 8 * hi;
            size_t idx = ((size_t)b * Tn + t) * Dn + d;
            float xp = x[idx] - ((d == 0) ? 1.f : 0.f) + c[i] + bias;
            xprime[idx] = xp;
            xpb[idx] = (bf16_t)xp;
        }
    }
}

// ------------------------------------------------------- Y = X' V  (WMMA)
__global__ __launch_bounds__(256) void k_xv(
    const bf16_t* __restrict__ xb, const bf16_t* __restrict__ Vt, float* __restrict__ out)
{
    int lane = threadIdx.x & 31, w = threadIdx.x >> 5;
    int tile = blockIdx.x * 8 + w;
    int b = tile / (Tn / 16);
    int t0 = (tile % (Tn / 16)) * 16;
    int nrow = lane & 15, hi = lane >> 4;
    const bf16_t* arow = xb + ((size_t)b * Tn + t0 + nrow) * Dn;
    const bf16_t* brow = Vt + ((size_t)b * Rn + nrow) * Dn;
    v8f c = vzero8();
    for (int kk = 0; kk < Dn; kk += 32) {
        v16bf A = wmma_lda_bf(arow + kk, hi);
        v16bf Bv = *(const v16bf*)(brow + kk + hi * 16);
        c = wmma_bf16(A, Bv, c);
    }
#pragma unroll
    for (int i = 0; i < 8; ++i) {
        int t = t0 + i + 8 * hi;
        out[((size_t)b * Tn + t) * Rn + nrow] = c[i];
    }
}

// -------------------------------- Zpart[b][tc] = sum_{t in chunk} X'^T Y
__global__ __launch_bounds__(256) void k_cov(
    const bf16_t* __restrict__ xb, const float* __restrict__ Y, float* __restrict__ Zpart)
{
    __shared__ float Ys[512 * Rn];  // 32KB
    int b = blockIdx.z, tc = blockIdx.y;
    int d = blockIdx.x * 256 + threadIdx.x;
    int t0 = tc * 512;
    for (int i = threadIdx.x; i < 512 * Rn; i += 256)
        Ys[i] = Y[((size_t)b * Tn + t0 + i / Rn) * Rn + (i % Rn)];
    __syncthreads();
    float acc[Rn];
#pragma unroll
    for (int r = 0; r < Rn; ++r) acc[r] = 0.f;
    for (int t = 0; t < 512; ++t) {
        float xv = (float)xb[((size_t)b * Tn + t0 + t) * Dn + d];
        const float* yr = &Ys[t * Rn];
#pragma unroll
        for (int r = 0; r < Rn; ++r) acc[r] = fmaf(xv, yr[r], acc[r]);
    }
#pragma unroll
    for (int r = 0; r < Rn; ++r)
        Zpart[(((size_t)b * 8 + tc) * Dn + d) * Rn + r] = acc[r];
}

// -------------------- reduce partials, +eps*V, modified Gram-Schmidt (QR)
__global__ __launch_bounds__(256) void k_orth(
    const float* __restrict__ Zpart, float* __restrict__ V, bf16_t* __restrict__ Vt,
    int apply_sign)
{
    __shared__ float red[8];
    __shared__ float sgn[16];
    int b = blockIdx.x, tid = threadIdx.x;
    float z[4][Rn];
#pragma unroll
    for (int q = 0; q < 4; ++q) {
        int d = tid * 4 + q;
#pragma unroll
        for (int r = 0; r < Rn; ++r) {
            float a = 0.f;
            for (int c = 0; c < 8; ++c)
                a += Zpart[(((size_t)b * 8 + c) * Dn + d) * Rn + r];
            z[q][r] = a * (1.f / Tn) + EPSc * V[((size_t)b * Dn + d) * Rn + r];
        }
    }
    __syncthreads();
    for (int j = 0; j < Rn; ++j) {
        float nv = 0.f;
#pragma unroll
        for (int q = 0; q < 4; ++q) nv += z[q][j] * z[q][j];
        float nrm = sqrtf(block_sum(nv, red));
        float inv = 1.f / fmaxf(nrm, 1e-30f);
#pragma unroll
        for (int q = 0; q < 4; ++q) z[q][j] *= inv;
        for (int i = j + 1; i < Rn; ++i) {
            float dv = 0.f;
#pragma unroll
            for (int q = 0; q < 4; ++q) dv += z[q][j] * z[q][i];
            float dot = block_sum(dv, red);
#pragma unroll
            for (int q = 0; q < 4; ++q) z[q][i] -= dot * z[q][j];
        }
    }
    if (apply_sign) {
        if (tid == 0) {  // thread 0 owns d=0 (anchor row)
#pragma unroll
            for (int r = 0; r < Rn; ++r)
                sgn[r] = ((z[0][r] + 1e-12f) >= 0.f) ? 1.f : -1.f;
        }
        __syncthreads();
#pragma unroll
        for (int q = 0; q < 4; ++q)
#pragma unroll
            for (int r = 0; r < Rn; ++r) z[q][r] *= sgn[r];
    }
#pragma unroll
    for (int q = 0; q < 4; ++q) {
        int d = tid * 4 + q;
#pragma unroll
        for (int r = 0; r < Rn; ++r) {
            float v = z[q][r];
            V[((size_t)b * Dn + d) * Rn + r] = v;
            Vt[((size_t)b * Rn + r) * Dn + d] = (bf16_t)v;
        }
    }
}

// --------------------------------------- scales[b][r] = sqrt(sum_t tr^2+eps)
__global__ __launch_bounds__(256) void k_scales(
    const float* __restrict__ traces, float* __restrict__ scales)
{
    __shared__ float red[8];
    int b = blockIdx.x, tid = threadIdx.x;
    float acc[Rn];
#pragma unroll
    for (int r = 0; r < Rn; ++r) acc[r] = 0.f;
    for (int t = tid; t < Tn; t += 256) {
        const float* p = traces + ((size_t)b * Tn + t) * Rn;
#pragma unroll
        for (int r = 0; r < Rn; ++r) { float v = p[r]; acc[r] = fmaf(v, v, acc[r]); }
    }
    for (int r = 0; r < Rn; ++r) {
        float s = block_sum(acc[r], red);
        if (tid == 0) scales[b * Rn + r] = sqrtf(s + EPSc);
    }
}

// ----- shrink + x_tilde = tf V^T; xhat = x_tilde + x - xprime  (bf16 out)
__global__ __launch_bounds__(256) void k_mix(
    const float* __restrict__ traces, const float* __restrict__ scales,
    const float* __restrict__ V, const float* __restrict__ x,
    const float* __restrict__ xprime, bf16_t* __restrict__ xhb)
{
    __shared__ float tfs[16][Rn];
    int tile = blockIdx.x;
    int b = tile / (Tn / 16);
    int t0 = (tile % (Tn / 16)) * 16;
    int tid = threadIdx.x;
    {
        int tt = tid >> 4, r = tid & 15;
        float tr = traces[((size_t)b * Tn + t0 + tt) * Rn + r];
        float sc = scales[b * Rn + r];
        tfs[tt][r] = soft_shrink(tr / sc) * sc;
    }
    __syncthreads();
    for (int d = tid; d < Dn; d += 256) {
        const float* vp = V + ((size_t)b * Dn + d) * Rn;
        float vr[Rn];
#pragma unroll
        for (int r = 0; r < Rn; ++r) vr[r] = vp[r];
#pragma unroll
        for (int tt = 0; tt < 16; ++tt) {
            float dotv = 0.f;
#pragma unroll
            for (int r = 0; r < Rn; ++r) dotv = fmaf(tfs[tt][r], vr[r], dotv);
            size_t idx = ((size_t)b * Tn + t0 + tt) * Dn + d;
            xhb[idx] = (bf16_t)(dotv + x[idx] - xprime[idx]);
        }
    }
}

// ----------------------------- y = x + xhat @ outW^T  (big WMMA GEMM)
// A panel (16 rows of xhat, contiguous 32KB) is staged into LDS once per block
// via gfx1250 async Global->LDS DMA and shared by all 8 waves.
__global__ __launch_bounds__(256) void k_gemm_out(
    const bf16_t* __restrict__ xhb, const bf16_t* __restrict__ Wb,
    const float* __restrict__ x, float* __restrict__ y)
{
    __shared__ bf16_t As[16 * Dn];  // 32KB
    int lane = threadIdx.x & 31, w = threadIdx.x >> 5;
    int tile = blockIdx.x;
    int b = tile / (Tn / 16);
    int t0 = (tile % (Tn / 16)) * 16;
    int n0 = (blockIdx.y * 8 + w) * 16;
    int nrow = lane & 15, hi = lane >> 4;

    // cooperative async stage: 2048 x 16B chunks across 256 threads
    {
        const char* gbase = (const char*)(xhb + ((size_t)b * Tn + t0) * Dn);
        unsigned lbase = lds_off32(As);
#pragma unroll
        for (int c = 0; c < 8; ++c) {
            int i = threadIdx.x + c * 256;
            async_copy_b128(lbase + (unsigned)i * 16, gbase + (size_t)i * 16);
        }
        wait_async0();
    }
    __syncthreads();

    const bf16_t* arow = As + nrow * Dn;
    const bf16_t* brow = Wb + (size_t)(n0 + nrow) * Dn;
    v8f c = vzero8();
    for (int kk = 0; kk < Dn; kk += 32) {
        __builtin_prefetch(brow + kk + 256, 0, 0);   // global_prefetch_b8
        v16bf A = wmma_lda_bf(arow + kk, hi);
        v16bf Bv = *(const v16bf*)(brow + kk + hi * 16);
        c = wmma_bf16(A, Bv, c);
    }
    int e = n0 + nrow;
#pragma unroll
    for (int i = 0; i < 8; ++i) {
        int t = t0 + i + 8 * hi;
        size_t idx = ((size_t)b * Tn + t) * Dn + e;
        y[idx] = c[i] + x[idx];
    }
}

// ------------------------------------------------------ LayerNorm in place
__global__ __launch_bounds__(256) void k_ln(
    float* __restrict__ y, const float* __restrict__ lw, const float* __restrict__ lb)
{
    __shared__ float red[8];
    size_t row = blockIdx.x;
    float* p = y + row * Dn;
    int tid = threadIdx.x;
    v4f v = *(const v4f*)(p + tid * 4);
    float s = v[0] + v[1] + v[2] + v[3];
    float mu = block_sum(s, red) * (1.f / Dn);
    float sq = 0.f;
#pragma unroll
    for (int i = 0; i < 4; ++i) { float d = v[i] - mu; sq += d * d; }
    float var = block_sum(sq, red) * (1.f / Dn);
    float inv = rsqrtf(var + LNEPSc);
    v4f wv = *(const v4f*)(lw + tid * 4);
    v4f bv = *(const v4f*)(lb + tid * 4);
    v4f o;
#pragma unroll
    for (int i = 0; i < 4; ++i) o[i] = (v[i] - mu) * inv * wv[i] + bv[i];
    *(v4f*)(p + tid * 4) = o;
}

// ===========================================================================
extern "C" void kernel_launch(void* const* d_in, const int* in_sizes, int n_in,
                              void* d_out, int out_size, void* d_ws, size_t ws_size,
                              hipStream_t stream)
{
    (void)in_sizes; (void)n_in; (void)out_size; (void)ws_size;
    const float* x    = (const float*)d_in[0];
    const float* Win  = (const float*)d_in[1];
    const float* Wout = (const float*)d_in[2];
    const float* sob  = (const float*)d_in[3];
    const float* outW = (const float*)d_in[4];
    const float* lw   = (const float*)d_in[5];
    const float* lb   = (const float*)d_in[6];
    float* y = (float*)d_out;

    char* ws = (char*)d_ws;
    size_t off = 0;
    auto carve = [&](size_t bytes) -> void* {
        void* p = ws + off;
        off += (bytes + 255) & ~(size_t)255;
        return p;
    };
    bf16_t* xhb    = (bf16_t*)carve((size_t)Bn * Tn * Dn * sizeof(bf16_t)); // X' bf16, later xhat bf16
    bf16_t* Winb   = (bf16_t*)carve((size_t)SRn * Dn * sizeof(bf16_t));
    bf16_t* Woutb  = (bf16_t*)carve((size_t)Dn * SRn * sizeof(bf16_t));
    bf16_t* outWb  = (bf16_t*)carve((size_t)Dn * Dn * sizeof(bf16_t));
    float*  V      = (float*)carve((size_t)Bn * Dn * Rn * sizeof(float));
    bf16_t* Vt     = (bf16_t*)carve((size_t)Bn * Rn * Dn * sizeof(bf16_t));
    float*  Ybuf   = (float*)carve((size_t)Bn * Tn * Rn * sizeof(float));  // Y, then traces
    float*  Zp     = (float*)carve((size_t)Bn * 8 * Dn * Rn * sizeof(float));
    float*  scales = (float*)carve((size_t)Bn * Rn * sizeof(float));
    float*  xprime = y;  // stage xprime fp32 in d_out; overwritten by final GEMM

    k_cvt_bf16<<<(SRn * Dn + 255) / 256, 256, 0, stream>>>(Win, Winb, SRn * Dn);
    k_cvt_bf16<<<(Dn * SRn + 255) / 256, 256, 0, stream>>>(Wout, Woutb, Dn * SRn);
    k_cvt_bf16<<<(Dn * Dn + 255) / 256, 256, 0, stream>>>(outW, outWb, Dn * Dn);
    k_init_V<<<(Bn * Dn * Rn + 255) / 256, 256, 0, stream>>>(V, Vt);

    k_shift<<<Bn * (Tn / 16) / 8, 256, 0, stream>>>(x, Winb, Woutb, sob, xprime, xhb);

    for (int k = 0; k < KIT; ++k) {
        k_xv<<<Bn * (Tn / 16) / 8, 256, 0, stream>>>(xhb, Vt, Ybuf);
        k_cov<<<dim3(Dn / 256, 8, Bn), 256, 0, stream>>>(xhb, Ybuf, Zp);
        k_orth<<<Bn, 256, 0, stream>>>(Zp, V, Vt, (k == KIT - 1) ? 1 : 0);
    }

    k_xv<<<Bn * (Tn / 16) / 8, 256, 0, stream>>>(xhb, Vt, Ybuf);          // traces
    k_scales<<<Bn, 256, 0, stream>>>(Ybuf, scales);
    k_mix<<<Bn * (Tn / 16), 256, 0, stream>>>(Ybuf, scales, V, x, xprime, xhb);
    k_gemm_out<<<dim3(Bn * (Tn / 16), Dn / 16 / 8), 256, 0, stream>>>(xhb, outWb, x, y);
    k_ln<<<Bn * Tn, 256, 0, stream>>>(y, lw, lb);
}
